// density_wfa_31190052503758
// MI455X (gfx1250) — matrix operand order repaired and checked by reading, not compiled, not run on hardware
//
#include <hip/hip_runtime.h>
#include <hip/hip_bf16.h>

typedef __attribute__((ext_vector_type(16))) _Float16 v16h;
typedef __attribute__((ext_vector_type(8)))  _Float16 v8h;
typedef __attribute__((ext_vector_type(8)))  float    v8f;

union U16 { v16h v; v8h h[2]; };

#define NBLK   16
#define ROWS   128            // batch rows per block
#define B_TOT  2048
#define L_SEQ  128
#define NSTEP  128
#define RDIM   64
#define MD     256            // MIXN*XD
#define XD     8
#define EPSBN  1e-5f
#define LOG2PI 1.8378770664093453f

struct P {
    const float *X, *enc_w, *enc_b, *init_w, *bn_g, *bn_b;
    const float *mub, *mub2, *sigb, *sigb2, *alb, *alb2;
    const _Float16 *Apack, *W1mu, *W1sig, *W1al, *W2mu, *W2sig, *W2al;
    _Float16 *s1mu, *s1sig;
    float *muws, *sigws;
    unsigned *cnt, *gen;
    float *gNorm, *gSum, *gSqs;
    float *out;
};

__device__ __forceinline__ v8f wmma16(v16h a, v16h b, v8f c) {
    return __builtin_amdgcn_wmma_f32_16x16x32_f16(false, a, false, b, (short)0, c,
                                                  false, false);
}

__device__ __forceinline__ float gload(const float* p) {
    return __hip_atomic_load(p, __ATOMIC_RELAXED, __HIP_MEMORY_SCOPE_AGENT);
}

// 16-byte async copy global -> LDS (ASYNCcnt-tracked, ISA 15.18.3 opcode 98).
__device__ __forceinline__ void async_cp16(const void* g, void* l) {
    unsigned loff = (unsigned)(size_t)(__attribute__((address_space(3))) char*)l;
    unsigned long long ga = (unsigned long long)(size_t)g;
    asm volatile("global_load_async_to_lds_b128 %0, %1, off"
                 :: "v"(loff), "v"(ga) : "memory");
}

// Grid barrier: cluster barrier (NOP when unclustered; HW barrier when the 16
// workgroups are dispatched as one cluster) + atomic generation barrier.
__device__ __forceinline__ void gbar(unsigned* cnt, unsigned* gen) {
    __syncthreads();
    if (threadIdx.x == 0) {
        __builtin_amdgcn_s_cluster_barrier();   // s_barrier_signal/wait -3
        __threadfence();
        unsigned g = __hip_atomic_load(gen, __ATOMIC_RELAXED, __HIP_MEMORY_SCOPE_AGENT);
        unsigned a = __hip_atomic_fetch_add(cnt, 1u, __ATOMIC_ACQ_REL,
                                            __HIP_MEMORY_SCOPE_AGENT);
        if (a == NBLK - 1) {
            __hip_atomic_store(cnt, 0u, __ATOMIC_RELAXED, __HIP_MEMORY_SCOPE_AGENT);
            __hip_atomic_fetch_add(gen, 1u, __ATOMIC_RELEASE, __HIP_MEMORY_SCOPE_AGENT);
        } else {
            while (__hip_atomic_load(gen, __ATOMIC_ACQUIRE,
                                     __HIP_MEMORY_SCOPE_AGENT) == g)
                __builtin_amdgcn_s_sleep(1);
        }
        __threadfence();
    }
    __syncthreads();
}

// Pack a K x N weight (f32) into WMMA B-fragments (f16).
// Fragment layout: dst[((kb*JT + jt)*32 + lane)*16 + e],
// element e of lane l = W[kb*32 + (e&7) + (e>=8 ? 16 : 0) + (l>=16 ? 8 : 0)][jt*16 + (l&15)].
// srcKN==1: src is row-major [K][N]; srcKN==0: src is [N][K] (torch Linear weight).
__global__ void pack_b(const float* __restrict__ src, _Float16* __restrict__ dst,
                       int K, int N, int srcKN) {
    int idx = blockIdx.x * 256 + threadIdx.x;
    int total = K * N;
    if (idx >= total) return;
    int e  = idx & 15;
    int l  = (idx >> 4) & 31;
    int f  = idx >> 9;
    int JT = N >> 4;
    int jt = f % JT;
    int kb = f / JT;
    int k  = kb * 32 + (e & 7) + ((e >= 8) ? 16 : 0) + ((l >= 16) ? 8 : 0);
    int n  = jt * 16 + (l & 15);
    float v = srcKN ? src[(size_t)k * N + n] : src[(size_t)n * K + k];
    dst[idx] = (_Float16)v;
}

__global__ void wfa_init(unsigned* cnt, unsigned* gen, float* gNorm,
                         float* gSum, float* gSqs) {
    int tid = threadIdx.x;
    if (tid == 0) { *cnt = 0u; *gen = 0u; *gNorm = 0.f; }
    for (int i = tid; i < NSTEP * RDIM; i += 256) { gSum[i] = 0.f; gSqs[i] = 0.f; }
}

// One 256-col MLP path: s1 = softsign(h@W1 + b1); out = s1@W2 + b2  (f32 to ws).
__device__ __forceinline__ void mlp_path(const _Float16* hRow,
                                         const _Float16* W1, const _Float16* W2,
                                         const float* b1, const float* b2,
                                         _Float16* s1, float* outw,
                                         int row0, int m0, int lm, int hl) {
    for (int jt = 0; jt < 16; ++jt) {
        v8f a = {};
#pragma unroll
        for (int kb = 0; kb < 2; ++kb) {
            U16 af;
            af.h[0] = *(const v8h*)(hRow + kb * 32 + hl * 8);
            af.h[1] = *(const v8h*)(hRow + kb * 32 + 16 + hl * 8);
            v16h bf = ((const v16h*)W1)[(size_t)(kb * 16 + jt) * 32 + (threadIdx.x & 31)];
            a = wmma16(af.v, bf, a);
        }
        int col = jt * 16 + lm;
        float bias = b1[col];
#pragma unroll
        for (int r = 0; r < 8; ++r) {
            float v = a[r] + bias;
            v = v / (1.f + fabsf(v));
            s1[(size_t)(row0 + m0 + r + 8 * hl) * MD + col] = (_Float16)v;
        }
    }
    asm volatile("s_wait_storecnt 0x0" ::: "memory");
    const _Float16* s1r = s1 + (size_t)(row0 + m0 + (threadIdx.x & 15)) * MD;
    for (int jt = 0; jt < 16; ++jt) {
        v8f a = {};
#pragma unroll 2
        for (int kb = 0; kb < 8; ++kb) {
            U16 af;
            af.h[0] = *(const v8h*)(s1r + kb * 32 + hl * 8);
            af.h[1] = *(const v8h*)(s1r + kb * 32 + 16 + hl * 8);
            v16h bf = ((const v16h*)W2)[(size_t)(kb * 16 + jt) * 32 + (threadIdx.x & 31)];
            a = wmma16(af.v, bf, a);
        }
        int col = jt * 16 + lm;
        float bias = b2[col];
#pragma unroll
        for (int r = 0; r < 8; ++r)
            outw[(size_t)(row0 + m0 + r + 8 * hl) * MD + col] = a[r] + bias;
    }
}

__global__ __launch_bounds__(256, 1) void wfa_main(P p) {
    __shared__ __align__(16) _Float16 sTmp[ROWS][RDIM];   // state (f16)
    __shared__ __align__(16) _Float16 sEnc[ROWS][RDIM];   // enc, then h
    // Phase-overlaid scratch (24 KB):
    //   trilinear phase: double-buffered async B-fragment staging 2 x 8 KB
    //   phi phase:       sA2 [128][32] f32 (16 KB) + sS1a [128][32] f16 (8 KB)
    __shared__ __align__(64) unsigned char sScr[24576];
    __shared__ float sRed[2][RDIM];
    __shared__ float sPair[ROWS][2];
    __shared__ float sRes[ROWS];

    float*    sA2f = (float*)sScr;               // [128][32]
    _Float16* sS1a = (_Float16*)(sScr + 16384);  // [128][32]

    const int tid  = threadIdx.x;
    const int wv   = tid >> 5;
    const int lane = tid & 31;
    const int lm   = lane & 15;
    const int hl   = lane >> 4;
    const int m0   = wv * 16;
    const int row0 = blockIdx.x * ROWS;

    for (int i = tid; i < ROWS; i += 256) sRes[i] = 0.f;
    float normAcc = 0.f;

    for (int t = 0; t < NSTEP; ++t) {
        __syncthreads();
        v8f acc[4] = {};

        if (t == 0) {
#pragma unroll
            for (int jt = 0; jt < 4; ++jt) {
                float w = p.init_w[jt * 16 + lm];
#pragma unroll
                for (int r = 0; r < 8; ++r) acc[jt][r] = w;
            }
        } else {
            // ---- encoder: enc = softsign(xp @ enc_w^T + enc_b) -> sEnc ----
            {
                const int row  = tid >> 1;
                const int half = tid & 1;
                const float* Xr = p.X + (size_t)(row0 + row) * (XD * L_SEQ) + (t - 1);
                float xp[XD];
#pragma unroll
                for (int k = 0; k < XD; ++k) {
                    xp[k] = Xr[k * L_SEQ];
                    __builtin_prefetch(Xr + k * L_SEQ + 1, 0, 0);  // next step's column
                }
                for (int c = half * 32; c < half * 32 + 32; ++c) {
                    float s = p.enc_b[c];
#pragma unroll
                    for (int k = 0; k < XD; ++k) s += xp[k] * p.enc_w[c * XD + k];
                    s = s / (1.f + fabsf(s));
                    sEnc[row][c] = (_Float16)s;
                }
            }
            __syncthreads();
            // ---- trilinear: new = (tmp (x) enc) @ Aflat, K = 4096 ----
            // B fragments stream through LDS via double-buffered async copies:
            // one 8 KB chunk (2 k-blocks) is consumed by all 8 waves while the
            // next is in flight (ASYNCcnt). 8x less L2 traffic than per-wave loads.
            const char* gA = (const char*)p.Apack;
            {
                const char* gs = gA + tid * 32;
                char* ls = (char*)sScr + tid * 32;
                async_cp16(gs, ls);
                async_cp16(gs + 16, ls + 16);
            }
            const _Float16* tmpRow = &sTmp[m0 + lm][0];
            const _Float16* encRow = &sEnc[m0 + lm][0];
            for (int c = 0; c < 64; ++c) {
                __syncthreads();   // everyone done reading the buffer being refilled
                if (c + 1 < 64) {
                    const char* gs = gA + (size_t)(c + 1) * 8192 + tid * 32;
                    char* ls = (char*)sScr + ((c + 1) & 1) * 8192 + tid * 32;
                    async_cp16(gs, ls);
                    async_cp16(gs + 16, ls + 16);
                    asm volatile("s_wait_asynccnt 0x2" ::: "memory");
                } else {
                    asm volatile("s_wait_asynccnt 0x0" ::: "memory");
                }
                __syncthreads();   // chunk c fully in LDS (all waves' portions)
                const char* bufB = (const char*)sScr + (c & 1) * 8192;
                _Float16 tv = tmpRow[c];          // i0 = kb>>1 = c for both k2
#pragma unroll
                for (int k2 = 0; k2 < 2; ++k2) {
                    const int d0 = k2 * 32;
                    U16 af;
                    af.h[0] = *(const v8h*)(encRow + d0 + hl * 8);
                    af.h[1] = *(const v8h*)(encRow + d0 + 16 + hl * 8);
                    af.v = af.v * tv;
                    const v16h* bp = (const v16h*)(bufB + k2 * 4096);
#pragma unroll
                    for (int jt = 0; jt < 4; ++jt)
                        acc[jt] = wmma16(af.v, bp[jt * 32 + lane], acc[jt]);
                }
            }
        }

        // ---- batch norm (grid-wide statistics) ----
        __syncthreads();
        if (tid < RDIM) { sRed[0][tid] = 0.f; sRed[1][tid] = 0.f; }
        __syncthreads();
#pragma unroll
        for (int jt = 0; jt < 4; ++jt) {
            float s = 0.f, q = 0.f;
#pragma unroll
            for (int r = 0; r < 8; ++r) { float v = acc[jt][r]; s += v; q += v * v; }
            atomicAdd(&sRed[0][jt * 16 + lm], s);
            atomicAdd(&sRed[1][jt * 16 + lm], q);
        }
        __syncthreads();
        if (tid < RDIM) {
            atomicAdd(&p.gSum[t * RDIM + tid], sRed[0][tid]);
            atomicAdd(&p.gSqs[t * RDIM + tid], sRed[1][tid]);
            __threadfence();
        }
        gbar(p.cnt, p.gen);
#pragma unroll
        for (int jt = 0; jt < 4; ++jt) {
            const int col = jt * 16 + lm;
            float mean = gload(&p.gSum[t * RDIM + col]) * (1.f / B_TOT);
            float var  = gload(&p.gSqs[t * RDIM + col]) * (1.f / B_TOT) - mean * mean;
            float scal = p.bn_g[col] * rsqrtf(var + EPSBN);
            float shft = p.bn_b[col] - mean * scal;
#pragma unroll
            for (int r = 0; r < 8; ++r) {
                float v = acc[jt][r] * scal + shft;
                normAcc += v * v;
                sTmp[m0 + r + 8 * hl][col] = (_Float16)v;
            }
        }
        __syncthreads();

        // ---- h = softmax(tmp) -> sEnc ----
        if (tid < ROWS) {
            float mx = -1e30f;
            for (int j = 0; j < RDIM; ++j) mx = fmaxf(mx, (float)sTmp[tid][j]);
            float sum = 0.f;
            for (int j = 0; j < RDIM; ++j) sum += __expf((float)sTmp[tid][j] - mx);
            float inv = 1.f / sum;
            for (int j = 0; j < RDIM; ++j)
                sEnc[tid][j] = (_Float16)(__expf((float)sTmp[tid][j] - mx) * inv);
        }
        __syncthreads();

        const _Float16* hRow = &sEnc[m0 + lm][0];

        // ---- mu and sigma MLP paths (WMMA) ----
        mlp_path(hRow, p.W1mu,  p.W2mu,  p.mub,  p.mub2,  p.s1mu,  p.muws,
                 row0, m0, lm, hl);
        mlp_path(hRow, p.W1sig, p.W2sig, p.sigb, p.sigb2, p.s1sig, p.sigws,
                 row0, m0, lm, hl);

        // ---- alpha path (WMMA, through LDS) ----
        for (int jt = 0; jt < 2; ++jt) {
            v8f a = {};
#pragma unroll
            for (int kb = 0; kb < 2; ++kb) {
                U16 af;
                af.h[0] = *(const v8h*)(hRow + kb * 32 + hl * 8);
                af.h[1] = *(const v8h*)(hRow + kb * 32 + 16 + hl * 8);
                v16h bf = ((const v16h*)p.W1al)[(size_t)(kb * 2 + jt) * 32 + lane];
                a = wmma16(af.v, bf, a);
            }
            int col = jt * 16 + lm;
            float bias = p.alb[col];
#pragma unroll
            for (int r = 0; r < 8; ++r) {
                float v = a[r] + bias;
                v = v / (1.f + fabsf(v));
                sS1a[(m0 + r + 8 * hl) * 32 + col] = (_Float16)v;
            }
        }
        __syncthreads();
        {
            const _Float16* aRow = sS1a + (m0 + lm) * 32;
            for (int jt = 0; jt < 2; ++jt) {
                v8f a = {};
                U16 af;
                af.h[0] = *(const v8h*)(aRow + hl * 8);
                af.h[1] = *(const v8h*)(aRow + 16 + hl * 8);
                v16h bf = ((const v16h*)p.W2al)[(size_t)jt * 32 + lane];
                a = wmma16(af.v, bf, a);
                int col = jt * 16 + lm;
                float bias = p.alb2[col];
#pragma unroll
                for (int r = 0; r < 8; ++r)
                    sA2f[(m0 + r + 8 * hl) * 32 + col] = a[r] + bias;
            }
        }
        asm volatile("s_wait_storecnt 0x0" ::: "memory");
        __threadfence_block();
        __syncthreads();

        // ---- GMM log-likelihood tail (2 threads per row, 16 mixtures each) ----
        {
            const int row  = tid >> 1;
            const int half = tid & 1;
            const float* Xr = p.X + (size_t)(row0 + row) * (XD * L_SEQ) + t;
            float x[XD];
#pragma unroll
            for (int k = 0; k < XD; ++k) x[k] = Xr[k * L_SEQ];
            // logsumexp over raw alpha logits
            float amax = -1e30f;
            for (int m = 0; m < 16; ++m) amax = fmaxf(amax, sA2f[row * 32 + half * 16 + m]);
            sPair[row][half] = amax; __syncthreads();
            amax = fmaxf(sPair[row][0], sPair[row][1]); __syncthreads();
            float asum = 0.f;
            for (int m = 0; m < 16; ++m) asum += __expf(sA2f[row * 32 + half * 16 + m] - amax);
            sPair[row][half] = asum; __syncthreads();
            float lseA = amax + __logf(sPair[row][0] + sPair[row][1]); __syncthreads();
            // u[mix] = alpha_logit + component log-prob
            float u[16];
            float umax = -1e30f;
            const float* muP = p.muws  + (size_t)(row0 + row) * MD + half * 16 * XD;
            const float* lsP = p.sigws + (size_t)(row0 + row) * MD + half * 16 * XD;
            for (int m = 0; m < 16; ++m) {
                float clp = 0.f;
#pragma unroll
                for (int k = 0; k < XD; ++k) {
                    float ls = lsP[m * XD + k];
                    float z  = (x[k] - muP[m * XD + k]) * __expf(-ls);
                    clp += -0.5f * z * z - ls;
                }
                clp -= 0.5f * XD * LOG2PI;
                u[m] = sA2f[row * 32 + half * 16 + m] + clp;
                umax = fmaxf(umax, u[m]);
            }
            sPair[row][half] = umax; __syncthreads();
            umax = fmaxf(sPair[row][0], sPair[row][1]); __syncthreads();
            float usum = 0.f;
            for (int m = 0; m < 16; ++m) usum += __expf(u[m] - umax);
            sPair[row][half] = usum; __syncthreads();
            if (half == 0)
                sRes[row] += (umax + __logf(sPair[row][0] + sPair[row][1])) - lseA;
        }
    }

    // ---- finalize ----
    __syncthreads();
    if (tid == 0) sRed[0][0] = 0.f;
    __syncthreads();
    atomicAdd(&sRed[0][0], normAcc);
    __syncthreads();
    if (tid == 0) { atomicAdd(p.gNorm, sRed[0][0]); __threadfence(); }
    for (int i = tid; i < ROWS; i += 256) p.out[row0 + i] = sRes[i];
    gbar(p.cnt, p.gen);
    if (blockIdx.x == 0 && tid == 0) p.out[B_TOT] = gload(p.gNorm);
}

extern "C" void kernel_launch(void* const* d_in, const int* in_sizes, int n_in,
                              void* d_out, int out_size, void* d_ws, size_t ws_size,
                              hipStream_t stream) {
    const float* X      = (const float*)d_in[0];
    const float* enc_w  = (const float*)d_in[1];
    const float* enc_b  = (const float*)d_in[2];
    const float* init_w = (const float*)d_in[3];
    const float* A      = (const float*)d_in[4];
    const float* bn_g   = (const float*)d_in[5];
    const float* bn_b   = (const float*)d_in[6];
    const float* muW    = (const float*)d_in[7];
    const float* mub    = (const float*)d_in[8];
    const float* muW2   = (const float*)d_in[9];
    const float* mub2   = (const float*)d_in[10];
    const float* sigW   = (const float*)d_in[11];
    const float* sigb   = (const float*)d_in[12];
    const float* sigW2  = (const float*)d_in[13];
    const float* sigb2  = (const float*)d_in[14];
    const float* alW    = (const float*)d_in[15];
    const float* alb    = (const float*)d_in[16];
    const float* alW2   = (const float*)d_in[17];
    const float* alb2   = (const float*)d_in[18];
    float* out = (float*)d_out;

    char* base = (char*)d_ws;
    size_t off = 0;
    auto alloc = [&](size_t bytes) -> void* {
        void* r = base + off;
        off = (off + bytes + 255) & ~(size_t)255;
        return r;
    };
    unsigned* cnt   = (unsigned*)alloc(16);
    unsigned* gen   = (unsigned*)alloc(16);
    float*    gNorm = (float*)   alloc(16);
    float*    gSum  = (float*)   alloc(NSTEP * RDIM * sizeof(float));
    float*    gSqs  = (float*)   alloc(NSTEP * RDIM * sizeof(float));
    float*    muws  = (float*)   alloc((size_t)B_TOT * MD * sizeof(float));
    float*    sigws = (float*)   alloc((size_t)B_TOT * MD * sizeof(float));
    _Float16* s1mu  = (_Float16*)alloc((size_t)B_TOT * MD * sizeof(_Float16));
    _Float16* s1sig = (_Float16*)alloc((size_t)B_TOT * MD * sizeof(_Float16));
    _Float16* Apack = (_Float16*)alloc((size_t)4096 * 64 * sizeof(_Float16));
    _Float16* W1mu  = (_Float16*)alloc((size_t)64 * 256 * sizeof(_Float16));
    _Float16* W1sig = (_Float16*)alloc((size_t)64 * 256 * sizeof(_Float16));
    _Float16* W1al  = (_Float16*)alloc((size_t)64 * 32 * sizeof(_Float16));
    _Float16* W2mu  = (_Float16*)alloc((size_t)256 * 256 * sizeof(_Float16));
    _Float16* W2sig = (_Float16*)alloc((size_t)256 * 256 * sizeof(_Float16));
    _Float16* W2al  = (_Float16*)alloc((size_t)32 * 32 * sizeof(_Float16));
    (void)ws_size; (void)in_sizes; (void)n_in; (void)out_size;

    wfa_init<<<1, 256, 0, stream>>>(cnt, gen, gNorm, gSum, gSqs);

    pack_b<<<(4096 * 64 + 255) / 256, 256, 0, stream>>>(A,     Apack, 4096, 64, 1);
    pack_b<<<(64 * 256 + 255) / 256,  256, 0, stream>>>(muW,   W1mu,  64, 256, 0);
    pack_b<<<(64 * 256 + 255) / 256,  256, 0, stream>>>(sigW,  W1sig, 64, 256, 0);
    pack_b<<<(64 * 32 + 255) / 256,   256, 0, stream>>>(alW,   W1al,  64, 32, 0);
    pack_b<<<(256 * 256 + 255) / 256, 256, 0, stream>>>(muW2,  W2mu,  256, 256, 0);
    pack_b<<<(256 * 256 + 255) / 256, 256, 0, stream>>>(sigW2, W2sig, 256, 256, 0);
    pack_b<<<(32 * 32 + 255) / 256,   256, 0, stream>>>(alW2,  W2al,  32, 32, 0);

    P p;
    p.X = X; p.enc_w = enc_w; p.enc_b = enc_b; p.init_w = init_w;
    p.bn_g = bn_g; p.bn_b = bn_b;
    p.mub = mub; p.mub2 = mub2; p.sigb = sigb; p.sigb2 = sigb2;
    p.alb = alb; p.alb2 = alb2;
    p.Apack = Apack; p.W1mu = W1mu; p.W1sig = W1sig; p.W1al = W1al;
    p.W2mu = W2mu; p.W2sig = W2sig; p.W2al = W2al;
    p.s1mu = s1mu; p.s1sig = s1sig; p.muws = muws; p.sigws = sigws;
    p.cnt = cnt; p.gen = gen; p.gNorm = gNorm; p.gSum = gSum; p.gSqs = gSqs;
    p.out = out;

    wfa_main<<<NBLK, 256, 0, stream>>>(p);
}
